// model_47888885350445
// MI455X (gfx1250) — matrix operand order repaired and checked by reading, not compiled, not run on hardware
//
#include <hip/hip_runtime.h>
#include <hip/hip_bf16.h>
#include <math.h>

// ---- problem constants (match reference) ----
#define Bsz 4096
#define Lsz 100
#define Esz 100
#define Hsz 64
#define H3  192       // 3*H
#define KP  128       // E padded to multiple of 32 for bf16 WMMA

typedef __attribute__((ext_vector_type(16))) __bf16 bf16x16;
typedef __attribute__((ext_vector_type(8)))  __bf16 bf16x8;
typedef __attribute__((ext_vector_type(8)))  float  f32x8;

union ABFrag { bf16x16 v; bf16x8 h[2]; };

// gfx1250 async global->LDS copy builtins (ASYNCcnt path).
#if __has_builtin(__builtin_amdgcn_global_load_async_to_lds_b128) && \
    __has_builtin(__builtin_amdgcn_s_wait_asynccnt)
#define ASYNC_LDS 1
typedef int v4i __attribute__((vector_size(16)));
typedef __attribute__((address_space(1))) v4i* as1_v4i_p;
typedef __attribute__((address_space(3))) v4i* as3_v4i_p;
#endif

static __device__ __forceinline__ f32x8 wmma_bf16(const bf16x16& a, const bf16x16& b, f32x8 c) {
  // (neg_a, A, neg_b, B, c_mod, C, reuse_a, reuse_b)
  return __builtin_amdgcn_wmma_f32_16x16x32_bf16(false, a, false, b, (short)0, c, false, false);
}

// =====================================================================
// Kernel A: TI[m, n] = x[m, :] @ W_in[:, n] + b_in[n]   for m in [0, B*L)
// Tiles: 64 rows (M) x 192 cols (N), K padded 100->128. 8 waves per block:
// wave = (nhalf<<2)|mstrip: mstrip picks 16 rows, nhalf picks 96 cols
// (6 N-tiles of 16). bf16 WMMA 16x16x32, f32 accumulate.
// =====================================================================
__global__ __launch_bounds__(256) void ti_gemm_kernel(
    const float* __restrict__ x, const float* __restrict__ Win,
    const float* __restrict__ bin, float* __restrict__ TI, int ntiles)
{
  __shared__ alignas(16) __bf16 Xs[64][KP];    // 16 KB
  __shared__ alignas(16) __bf16 Wcm[H3][KP];   // 48 KB, col-major (N-major), K zero-padded
  __shared__ float bsh[H3];

  const int tid = threadIdx.x;

  // one-time: W_in [E x 3H] row-major f32 -> col-major bf16, zero pad K
  for (int idx = tid; idx < H3 * KP; idx += 256) {
    int n = idx >> 7;          // / 128
    int k = idx & (KP - 1);
    float v = (k < Esz) ? Win[k * H3 + n] : 0.0f;
    Wcm[n][k] = (__bf16)v;
  }
  if (tid < H3) bsh[tid] = bin[tid];
  __syncthreads();

  const int lane  = tid & 31;
  const int wave  = tid >> 5;
  const int ms    = wave & 3;        // M strip (16 rows)
  const int nh    = wave >> 2;       // N half (96 cols)
  const int lrow  = lane & 15;
  const int lhalf = lane >> 4;       // 0 | 1

  for (int tile = blockIdx.x; tile < ntiles; tile += gridDim.x) {
    const long m0 = (long)tile * 64;

    // stage X tile -> bf16 LDS (zero padded K); conversion forces VGPR path
    for (int idx = tid; idx < 64 * KP; idx += 256) {
      int r = idx >> 7;
      int k = idx & (KP - 1);
      float v = (k < Esz) ? x[(m0 + r) * Esz + k] : 0.0f;
      Xs[r][k] = (__bf16)v;
    }
    __syncthreads();

    f32x8 acc[6] = {};
    #pragma unroll
    for (int kc = 0; kc < KP; kc += 32) {
      // A fragment (16x32): lanes 0-15 rows M, K chunks [kc+8h..] and [kc+16+8h..]
      ABFrag a;
      const __bf16* ap = &Xs[ms * 16 + lrow][kc + lhalf * 8];
      a.h[0] = *(const bf16x8*)ap;
      a.h[1] = *(const bf16x8*)(ap + 16);
      #pragma unroll
      for (int j = 0; j < 6; ++j) {
        // B fragment (32x16): col = lane%16, K chunks [kc+8h..] and [kc+16+8h..] from col-major LDS
        int n = nh * 96 + j * 16 + lrow;
        ABFrag b;
        const __bf16* bp = &Wcm[n][kc + lhalf * 8];
        b.h[0] = *(const bf16x8*)bp;
        b.h[1] = *(const bf16x8*)(bp + 16);
        acc[j] = wmma_bf16(a.v, b.v, acc[j]);
      }
    }

    // C layout: VGPR v -> row v + 8*lhalf, col lane%16. Fuse +b_in and store.
    #pragma unroll
    for (int j = 0; j < 6; ++j) {
      int nb = nh * 96 + j * 16 + lrow;
      float bv = bsh[nb];
      #pragma unroll
      for (int v = 0; v < 8; ++v) {
        long m = m0 + ms * 16 + v + lhalf * 8;
        TI[m * H3 + nb] = acc[j][v] + bv;
      }
    }
    __syncthreads();   // protect Xs before next tile overwrites it
  }
}

// =====================================================================
// Kernel B: per-block GRU recurrence over a 16-row batch tile, 2 phases
// (raw, standardized) x 100 steps, W_h resident in LDS as bf16 col-major,
// h kept in LDS (f32 + bf16). th = h @ W_h via bf16 WMMA (4 waves x 3
// N-tiles x K=64). TI tile staged with async global->LDS copies that
// overlap the WMMA GEMM. Final h @ W_fc + b_fc fused at the end.
// =====================================================================
__global__ __launch_bounds__(128) void gru_rec_kernel(
    const float* __restrict__ TI, const float* __restrict__ Wh,
    const float* __restrict__ bh, const float* __restrict__ Wfc,
    const float* __restrict__ bfc, float* __restrict__ out)
{
  __shared__ alignas(16) __bf16 Whcm[H3][Hsz];   // 24 KB col-major bf16
  __shared__ float bhs[H3];
  __shared__ alignas(16) float  hf[16][Hsz];     // h (f32)
  __shared__ alignas(16) __bf16 hb[16][Hsz];     // h (bf16, WMMA A operand)
  __shared__ alignas(16) float  tis[16][H3];     // input-projection tile
  __shared__ alignas(16) float  ths[16][H3];     // hidden-projection tile
  __shared__ float stats[16][4];                 // mean_i, istd_i, mean_h, istd_h

  const int tid = threadIdx.x;
  const int m0  = blockIdx.x * 16;   // batch rows [m0, m0+16)

  for (int idx = tid; idx < H3 * Hsz; idx += 128) {
    int n = idx >> 6;
    int k = idx & 63;
    Whcm[n][k] = (__bf16)Wh[k * H3 + n];
  }
  for (int idx = tid; idx < H3; idx += 128) bhs[idx] = bh[idx];
  for (int idx = tid; idx < 16 * Hsz; idx += 128) {
    int r = idx >> 6, j = idx & 63;
    hf[r][j] = 0.0f;
    hb[r][j] = (__bf16)0.0f;
  }

  const int lane  = tid & 31;
  const int wave  = tid >> 5;        // 4 waves, each owns 48 N columns
  const int lrow  = lane & 15;
  const int lhalf = lane >> 4;

  for (int phase = 0; phase < 2; ++phase) {
    for (int t = 0; t < Lsz; ++t) {
      __syncthreads();   // h from previous step visible; tis/ths free to reuse

      // stage this timestep's ti tile: 16 rows x 192 f32 = 768 x b128
#ifdef ASYNC_LDS
      #pragma unroll
      for (int i = 0; i < 6; ++i) {
        int chunk = tid + i * 128;        // 0..767
        int r = chunk / 48;
        int c4 = (chunk - r * 48) * 4;
        const float* g = &TI[((long)(m0 + r) * Lsz + t) * H3 + c4];
        __builtin_amdgcn_global_load_async_to_lds_b128(
            (as1_v4i_p)(void*)g,
            (as3_v4i_p)(void*)&tis[r][c4], 0, 0);
      }
#else
      for (int idx = tid; idx < 16 * H3; idx += 128) {
        int r = idx / H3;
        int n = idx - r * H3;
        tis[r][n] = TI[((long)(m0 + r) * Lsz + t) * H3 + n];
      }
#endif
      if (t + 1 < Lsz) {  // hint next timestep's tile toward the caches
        __builtin_prefetch(&TI[((long)(m0 + (tid >> 3)) * Lsz + t + 1) * H3 + (tid & 7) * 24], 0, 1);
      }

      // th = h @ W_h  (16x64 @ 64x192), K in two 32-chunks — overlaps async copy
      f32x8 acc[3] = {};
      #pragma unroll
      for (int kc = 0; kc < Hsz; kc += 32) {
        ABFrag a;
        const __bf16* ap = &hb[lrow][kc + lhalf * 8];
        a.h[0] = *(const bf16x8*)ap;
        a.h[1] = *(const bf16x8*)(ap + 16);
        #pragma unroll
        for (int j = 0; j < 3; ++j) {
          int n = wave * 48 + j * 16 + lrow;
          ABFrag b;
          const __bf16* bp = &Whcm[n][kc + lhalf * 8];
          b.h[0] = *(const bf16x8*)bp;
          b.h[1] = *(const bf16x8*)(bp + 16);
          acc[j] = wmma_bf16(a.v, b.v, acc[j]);
        }
      }
      #pragma unroll
      for (int j = 0; j < 3; ++j) {
        int nb = wave * 48 + j * 16 + lrow;
        float bv = bhs[nb];
        #pragma unroll
        for (int v = 0; v < 8; ++v)
          ths[v + lhalf * 8][nb] = acc[j][v] + bv;
      }
#ifdef ASYNC_LDS
      __builtin_amdgcn_s_wait_asynccnt(0);   // ti tile landed in LDS
#endif
      __syncthreads();   // tis + ths complete

      if (phase) {
        // per-row mean / unbiased std (ddof=1) over 192 features
        if (tid < 32) {
          int r = tid & 15;
          const float* src = (tid < 16) ? ths[r] : tis[r];
          float s = 0.0f, ss = 0.0f;
          for (int n = 0; n < H3; ++n) { float v = src[n]; s += v; ss += v * v; }
          float mean = s * (1.0f / (float)H3);
          float var  = (ss - (float)H3 * mean * mean) * (1.0f / (float)(H3 - 1));
          float istd = rsqrtf(var);
          if (tid < 16) { stats[r][2] = mean; stats[r][3] = istd; }
          else          { stats[r][0] = mean; stats[r][1] = istd; }
        }
        __syncthreads();
      }

      // GRU gates: 16 rows x 64 hidden = 1024 cells, 8 per thread
      #pragma unroll
      for (int i = 0; i < 8; ++i) {
        int idx = tid + i * 128;
        int r = idx >> 6, j = idx & 63;
        float ri = tis[r][j], zi = tis[r][64 + j], ni = tis[r][128 + j];
        float rh = ths[r][j], zh = ths[r][64 + j], nh2 = ths[r][128 + j];
        if (phase) {
          float mi = stats[r][0], ii = stats[r][1];
          float mh = stats[r][2], ih = stats[r][3];
          ri = (ri - mi) * ii;  zi = (zi - mi) * ii;  ni = (ni - mi) * ii;
          rh = (rh - mh) * ih;  zh = (zh - mh) * ih;  nh2 = (nh2 - mh) * ih;
        }
        float rt = 1.0f / (1.0f + __expf(-(ri + rh)));
        float zt = 1.0f / (1.0f + __expf(-(zi + zh)));
        float nt = tanhf(ni + rt * nh2);
        float hn = (1.0f - zt) * nt + zt * hf[r][j];
        hf[r][j] = hn;
        hb[r][j] = (__bf16)hn;
      }
    }
  }

  // final: out = h @ W_fc + b_fc  (16x64 @ 64x2)
  __syncthreads();
  if (tid < 32) {
    int r = tid >> 1, c = tid & 1;
    float s = bfc[c];
    for (int j = 0; j < Hsz; ++j) s += hf[r][j] * Wfc[j * 2 + c];
    out[(m0 + r) * 2 + c] = s;
  }
}

// =====================================================================
extern "C" void kernel_launch(void* const* d_in, const int* in_sizes, int n_in,
                              void* d_out, int out_size, void* d_ws, size_t ws_size,
                              hipStream_t stream) {
  const float* x    = (const float*)d_in[0];   // [4096,100,100]
  const float* Win  = (const float*)d_in[1];   // [100,192]
  const float* bin  = (const float*)d_in[2];   // [192]
  const float* Wh   = (const float*)d_in[3];   // [64,192]
  const float* bh   = (const float*)d_in[4];   // [192]
  const float* Wfc  = (const float*)d_in[5];   // [64,2]
  const float* bfc  = (const float*)d_in[6];   // [2]
  float* out = (float*)d_out;                  // [4096,2]

  float* TI = (float*)d_ws;                    // [B*L, 192] f32 (~315 MB)

  const int ntiles = (Bsz * Lsz) / 64;         // 6400 M-tiles of 64 rows
  ti_gemm_kernel<<<800, 256, 0, stream>>>(x, Win, bin, TI, ntiles);
  gru_rec_kernel<<<Bsz / 16, 128, 0, stream>>>(TI, Wh, bh, Wfc, bfc, out);
}
